// batchedGAT_66228395704911
// MI455X (gfx1250) — matrix operand order repaired and checked by reading, not compiled, not run on hardware
//
#include <hip/hip_runtime.h>

// ---------------- problem constants ----------------
constexpr int BB  = 16;     // batch
constexpr int CIN = 128;    // input channels (K of first GEMM)
constexpr int NN  = 2048;   // nodes
constexpr int FF  = 64;     // output features
constexpr float NEG_SLOPE = 0.33f;
constexpr float MASK_VAL  = -9.0e15f;

// ---------------- CDNA5 types ----------------
typedef __attribute__((ext_vector_type(16))) __bf16 v16bf;
typedef __attribute__((ext_vector_type(2)))  __bf16 v2bf;
typedef __attribute__((ext_vector_type(8)))  float  v8f;
typedef __attribute__((ext_vector_type(4)))  unsigned int v4u;
typedef __attribute__((ext_vector_type(8)))  int v8i;
typedef __attribute__((ext_vector_type(4)))  int v4i;

union BF16x16 { unsigned u[8]; v16bf v; };

__device__ __forceinline__ unsigned short f2bf(float f) {
  union { float f; unsigned u; } c; c.f = f;
  unsigned r = c.u + 0x7FFFu + ((c.u >> 16) & 1u);   // round-to-nearest-even
  return (unsigned short)(r >> 16);
}

#if defined(__has_builtin)
#if __has_builtin(__builtin_amdgcn_cvt_pk_bf16_f32)
#define HAVE_PK_BF16 1
#endif
#if __has_builtin(__builtin_amdgcn_tensor_load_to_lds)
#define HAVE_TDM 1
#endif
#endif
#ifndef HAVE_PK_BF16
#define HAVE_PK_BF16 0
#endif
#ifndef HAVE_TDM
#define HAVE_TDM 0
#endif

__device__ __forceinline__ unsigned pack_bf16(float lo, float hi) {
#if HAVE_PK_BF16
  union { v2bf v; unsigned u; } c;
  c.v = __builtin_amdgcn_cvt_pk_bf16_f32(lo, hi);    // 1 VALU op for 2 converts
  return c.u;
#else
  return (unsigned)f2bf(lo) | ((unsigned)f2bf(hi) << 16);
#endif
}

#if HAVE_TDM
// TDM: async-copy a 32-row x 64-col fp32 tile of g (row stride NN elements in
// memory) into LDS, padding 1 dword after every 64 dwords -> LDS row stride 65.
__device__ __forceinline__ void tdm_load_gtile(const float* gptr, unsigned ldsOff) {
  unsigned long long ga = (unsigned long long)(const void*)gptr;
  v4u d0;
  d0.x = 1u;                                                   // count=1 (valid D#)
  d0.y = ldsOff;                                               // lds_addr (bytes)
  d0.z = (unsigned)ga;                                         // global_addr[31:0]
  d0.w = ((unsigned)((ga >> 32) & 0x1FFFFFFull)) | (2u << 30); // addr[56:32] | type=2
  v8i d1;
  d1[0] = (2 << 16) | (1 << 20) | (5 << 22); // data_size=4B, pad_enable, interval=64dw, amount=1dw
  d1[1] = (int)(((unsigned)NN & 0xFFFFu) << 16);  // tensor_dim0[15:0] in bits[63:48]
  d1[2] = (int)(((unsigned)NN & 0xFFFFu) << 16);  // dim0 hi=0 | tensor_dim1[15:0]
  d1[3] = (64 << 16);                             // dim1 hi=0 | tile_dim0=64
  d1[4] = 32;                                     // tile_dim1=32 | tile_dim2=0
  d1[5] = NN;                                     // tensor_dim0_stride[31:0]
  d1[6] = 0;
  d1[7] = 0;
  v4i dz4 = {0, 0, 0, 0};
  v8i dz8 = {0, 0, 0, 0, 0, 0, 0, 0};
  // 6-arg form (amdgpu-toolchain / clang-23): (g0, g1, g2, g3, g4, cpol)
  __builtin_amdgcn_tensor_load_to_lds(d0, d1, dz4, dz4, dz8, 0);
}
#endif

// =====================================================================
// Kernel 1: Wh = h^T @ W   (bf16 WMMA, f32 accumulate)
//   - writes WhT as [b][f][n] bf16 (F-major) for the attention kernel
//   - computes f1 = Wh@a1, f2 = Wh@a2 via cross-lane reductions
// block = 128 threads (4 waves); each block: 16 nodes x all 64 features
// =====================================================================
__global__ __launch_bounds__(128) void gat_wh_kernel(
    const float* __restrict__ h, const float* __restrict__ W,
    const float* __restrict__ a, unsigned short* __restrict__ WhT,
    float* __restrict__ f1, float* __restrict__ f2)
{
  const int b  = blockIdx.y;
  const int i0 = blockIdx.x * 16;
  const int tid = threadIdx.x;
  const int wv = tid >> 5;
  const int L  = tid & 31;
  const int lm = L & 15;
  const int hi = (L >> 4) & 1;
  const int f0 = wv * 16;

  __shared__ float sF1[4][16];
  __shared__ float sF2[4][16];

  const float a1v = a[f0 + lm];
  const float a2v = a[FF + f0 + lm];

  v8f acc = {};
  #pragma unroll
  for (int s = 0; s < 4; ++s) {
    const int k0 = s * 32;
    BF16x16 A, Bm;
    #pragma unroll
    for (int v = 0; v < 8; ++v) {
      // A (16x32 bf16): lane lm = row m; dword v holds K pair per ISA layout
      int ka = k0 + ((v < 4) ? (2 * v) : (16 + 2 * (v - 4))) + (hi ? 8 : 0);
      float alo = h[((size_t)(b * CIN + ka)) * NN + i0 + lm];
      float ahi = h[((size_t)(b * CIN + ka + 1)) * NN + i0 + lm];
      A.u[v] = pack_bf16(alo, ahi);
      // B (32x16 bf16): lane lm = col f; dword v holds K pair per ISA layout (same as A)
      int kb = k0 + ((v < 4) ? (2 * v) : (16 + 2 * (v - 4))) + (hi ? 8 : 0);
      float blo = W[kb * FF + f0 + lm];
      float bhi = W[(kb + 1) * FF + f0 + lm];
      Bm.u[v] = pack_bf16(blo, bhi);
    }
    acc = __builtin_amdgcn_wmma_f32_16x16x32_bf16(
        false, A.v, false, Bm.v, (short)0, acc, false, false);
  }

  // store Wh transposed (F-major, bf16, dword-packed pairs of rows)
  {
    unsigned* dst = (unsigned*)(WhT + ((size_t)(b * FF) + f0 + lm) * NN + i0 + (hi ? 8 : 0));
    #pragma unroll
    for (int r = 0; r < 4; ++r) dst[r] = pack_bf16(acc[2 * r], acc[2 * r + 1]);
  }
  // f1/f2 partials via cross-lane reduction within each 16-lane half
  #pragma unroll
  for (int r = 0; r < 8; ++r) {
    const int M = r + (hi ? 8 : 0);
    const float val = acc[r];
    float s1 = val * a1v, s2 = val * a2v;
    #pragma unroll
    for (int m = 1; m < 16; m <<= 1) {
      s1 += __shfl_xor(s1, m, 32);
      s2 += __shfl_xor(s2, m, 32);
    }
    if (lm == 0) { sF1[wv][M] = s1; sF2[wv][M] = s2; }
  }
  __syncthreads();
  if (tid < 16) {
    float s1 = sF1[0][tid] + sF1[1][tid] + sF1[2][tid] + sF1[3][tid];
    float s2 = sF2[0][tid] + sF2[1][tid] + sF2[2][tid] + sF2[3][tid];
    f1[b * NN + i0 + tid] = s1;
    f2[b * NN + i0 + tid] = s2;
  }
}

// =====================================================================
// Kernel 2: fused masked-softmax attention + PV (flash style)
//   streams g (268 MB -> HBM bound, ~11.5us floor) via TDM double buffer
// block = 256 threads (8 waves) = 2 row-groups x 4 f-chunks; 32 rows/block
// =====================================================================
__global__ __launch_bounds__(256) void gat_attn_kernel(
    const float* __restrict__ g, const unsigned short* __restrict__ WhT,
    const float* __restrict__ f1, const float* __restrict__ f2,
    float* __restrict__ out)
{
  const int b  = blockIdx.y;
  const int i0 = blockIdx.x * 32;
  const int tid = threadIdx.x;
  const int wv = tid >> 5;
  const int L  = tid & 31;
  const int lm = L & 15;
  const int hi = (L >> 4) & 1;
  const int rg = wv >> 2;       // row group (0/1)
  const int f0 = (wv & 3) * 16; // feature chunk

  __shared__ float sG[2][32 * 65];            // g tiles, stride 65 (TDM pad)
  __shared__ unsigned short sV[64 * 72];      // V^T tile [f][j], stride 72
  __shared__ unsigned short sP[32 * 66];      // P tile bf16, stride 66
  __shared__ float sRowMax[32], sRowSum[32], sScale[32];

  const float* gb = g + (size_t)b * NN * NN;

  // e-phase mapping: 8 threads per row, 8 cols per thread
  const int er = tid >> 3;
  const int ec = (tid & 7) * 8;
  const float f1v = f1[b * NN + i0 + er];

  if (tid < 32) { sRowMax[tid] = -1e38f; sRowSum[tid] = 0.f; }

  v8f acc = {};

  const int T = NN / 64;
#if HAVE_TDM
  if (tid < 32)
    tdm_load_gtile(gb + (size_t)i0 * NN, (unsigned)(unsigned long long)(const void*)&sG[0][0]);
#endif

  for (int t = 0; t < T; ++t) {
    const int j0 = t * 64;
    int bufc = t & 1;
#if HAVE_TDM
    if (tid < 32) __builtin_amdgcn_s_wait_tensorcnt(0);
    __syncthreads();  // tile t visible; previous compute done
    if (tid < 32 && t + 1 < T)
      tdm_load_gtile(gb + (size_t)i0 * NN + (j0 + 64),
                     (unsigned)(unsigned long long)(const void*)&sG[bufc ^ 1][0]);
#else
    __syncthreads();
    {
      const float* src = gb + (size_t)i0 * NN + j0;
      #pragma unroll
      for (int k = 0; k < 8; ++k)
        sG[0][er * 65 + ec + k] = src[(size_t)er * NN + ec + k];
    }
    __syncthreads();
    bufc = 0;
#endif
    // stage V tile (64 f-rows x 64 j-cols of bf16 Wh^T) into LDS
    {
      const int fr = tid >> 2;
      const int cc = (tid & 3) * 16;
      const uint4* src = (const uint4*)(WhT + ((size_t)(b * FF) + fr) * NN + j0 + cc);
      uint4* dst = (uint4*)(sV + fr * 72 + cc);
      dst[0] = src[0];
      dst[1] = src[1];
    }

    // e, online-softmax update, P->bf16
    float ev[8];
    float tmax = -1e38f;
    #pragma unroll
    for (int k = 0; k < 8; ++k) {
      float gv = sG[bufc][er * 65 + ec + k];
      float x  = f1v + f2[b * NN + j0 + ec + k];
      float e  = (x > 0.f) ? x : NEG_SLOPE * x;
      e = (gv > 0.f) ? e : MASK_VAL;
      ev[k] = e;
      tmax = fmaxf(tmax, e);
    }
    #pragma unroll
    for (int m = 1; m < 8; m <<= 1) tmax = fmaxf(tmax, __shfl_xor(tmax, m, 32));
    const float mold = sRowMax[er];
    const float nm   = fmaxf(mold, tmax);
    const float scl  = __expf(mold - nm);
    float tsum = 0.f;
    unsigned* pdst = (unsigned*)&sP[er * 66 + ec];
    #pragma unroll
    for (int k = 0; k < 8; k += 2) {
      float p0 = __expf(ev[k] - nm);
      float p1 = __expf(ev[k + 1] - nm);
      tsum += p0 + p1;
      pdst[k >> 1] = pack_bf16(p0, p1);
    }
    #pragma unroll
    for (int m = 1; m < 8; m <<= 1) tsum += __shfl_xor(tsum, m, 32);
    if ((tid & 7) == 0) {
      sRowMax[er] = nm;
      sRowSum[er] = sRowSum[er] * scl + tsum;
      sScale[er]  = scl;
    }
    __syncthreads();  // sP, sV, sScale ready

    // rescale accumulator by exp(m_old - m_new) per row
    #pragma unroll
    for (int r = 0; r < 8; ++r) acc[r] *= sScale[rg * 16 + r + (hi ? 8 : 0)];

    // P (16x64) @ V (64x16) as two 16x16x32 bf16 WMMAs
    #pragma unroll
    for (int kb = 0; kb < 64; kb += 32) {
      BF16x16 A, Bm;
      #pragma unroll
      for (int v = 0; v < 8; ++v) {
        int ka = kb + ((v < 4) ? (2 * v) : (16 + 2 * (v - 4))) + (hi ? 8 : 0);
        A.u[v] = *(const unsigned*)&sP[(rg * 16 + lm) * 66 + ka];
        int kv = kb + ((v < 4) ? (2 * v) : (16 + 2 * (v - 4))) + (hi ? 8 : 0);
        Bm.u[v] = *(const unsigned*)&sV[(f0 + lm) * 72 + kv];
      }
      acc = __builtin_amdgcn_wmma_f32_16x16x32_bf16(
          false, A.v, false, Bm.v, (short)0, acc, false, false);
    }
  }

  // normalize and store transposed output: out[b][f][i]
  #pragma unroll
  for (int r = 0; r < 8; ++r) {
    const int row = rg * 16 + r + (hi ? 8 : 0);
    const float inv = 1.0f / sRowSum[row];
    out[((size_t)(b * FF) + f0 + lm) * NN + i0 + row] = acc[r] * inv;
  }
}

// =====================================================================
extern "C" void kernel_launch(void* const* d_in, const int* in_sizes, int n_in,
                              void* d_out, int out_size, void* d_ws, size_t ws_size,
                              hipStream_t stream) {
  const float* h = (const float*)d_in[0];   // (B, C, N, 1)
  const float* g = (const float*)d_in[1];   // (B, 1, N, N)
  const float* W = (const float*)d_in[2];   // (C, F)
  const float* a = (const float*)d_in[3];   // (2F, 1)
  float* out = (float*)d_out;               // (B, F, N, 1)

  unsigned short* WhT = (unsigned short*)d_ws;                       // B*F*N bf16 (4 MB)
  float* f1 = (float*)((char*)d_ws + (size_t)BB * FF * NN * 2);      // B*N f32
  float* f2 = f1 + (size_t)BB * NN;                                  // B*N f32

  gat_wh_kernel<<<dim3(NN / 16, BB), 128, 0, stream>>>(h, W, a, WhT, f1, f2);
  gat_attn_kernel<<<dim3(NN / 32, BB), 256, 0, stream>>>(g, WhT, f1, f2, out);
}
